// DynamicSparseAttention_13932873908680
// MI455X (gfx1250) — matrix operand order repaired and checked by reading, not compiled, not run on hardware
//
#include <hip/hip_runtime.h>
#include <cstdint>
#include <cstddef>

// ---------------------------------------------------------------------------
// DynamicSparseAttention for MI455X (gfx1250, wave32, WMMA).
// B=2, L=2048, D=1024, NH=16, KVH=8, DH=64, GROUPS=2, top_k = L/2 = 1024.
//
// All matrix math runs on v_wmma_f32_16x16x32_bf16 (fp32 accumulate).
// The 16x2048 fp32 score stripe per (b,h,qtile) lives entirely in LDS
// (128 KB < 320 KB/WGP) so the O(L^2) score tensor never hits HBM.
// ---------------------------------------------------------------------------

#define BB   2
#define LL   2048
#define DD   1024
#define NHH  16
#define KVHH 8
#define DHH  64
#define GRP  2
#define TOPK 1024
#define MTOT (BB * LL)            // 4096 token rows
#define NQKV 2048                 // Q(1024) | K(512) | V(512)

typedef __bf16 bf16;
typedef __attribute__((ext_vector_type(16))) __bf16 v16bf;
typedef __attribute__((ext_vector_type(8)))  __bf16 v8bf;
typedef __attribute__((ext_vector_type(8)))  float  v8f;

#define WMMA_BF16(a, b, c) \
  __builtin_amdgcn_wmma_f32_16x16x32_bf16(false, (a), false, (b), (short)0, (c), false, false)

// ---- WMMA fragment helpers (A/B share contiguous k = 16*(lane/16) + s; C/D per ISA) -------

// A (16x32 bf16, row-major source, stride lda elements):
// lanes 0-15 : M=lane,   halves 0..15 -> K k0+0..15
// lanes 16-31: M=lane-16, halves 0..15 -> K k0+16..31
__device__ __forceinline__ v16bf load_frag_a(const bf16* A, int lda, int row0,
                                             int k0, int lane) {
  const bf16* p = A + (size_t)(row0 + (lane & 15)) * lda + k0 + ((lane >> 4) << 4);
  v8bf lo = *(const v8bf*)p;         // 16B aligned (k0 % 32 == 0, lda % 8 == 0)
  v8bf hi = *(const v8bf*)(p + 8);
  return __builtin_shufflevector(lo, hi, 0, 1, 2, 3, 4, 5, 6, 7,
                                 8, 9, 10, 11, 12, 13, 14, 15);
}

// B (32x16 bf16) sourced from Bt = B^T (N x K row-major, stride ldb):
// lane n (0-15) holds column n0+n, K = k0..k0+15 ; lanes 16-31: K = k0+16..k0+31
__device__ __forceinline__ v16bf load_frag_b(const bf16* Bt, int ldb, int n0,
                                             int k0, int lane) {
  const bf16* p = Bt + (size_t)(n0 + (lane & 15)) * ldb + k0 + ((lane >> 4) << 4);
  return *(const v16bf*)p;           // 32B contiguous, 32B aligned
}

// C/D (16x16 f32): VGPR r, lanes 0-15 -> M=r, lanes 16-31 -> M=8+r; N = lane&15
__device__ __forceinline__ void store_tile_f32(float* C, int ldc, int m0, int n0,
                                               int lane, v8f c) {
  const int m = m0 + ((lane >> 4) << 3);
  const int n = n0 + (lane & 15);
#pragma unroll
  for (int r = 0; r < 8; ++r) C[(size_t)(m + r) * ldc + n] = c[r];
}

// ---- wave32 reductions ------------------------------------------------------
__device__ __forceinline__ float wred_max(float v) {
#pragma unroll
  for (int o = 16; o; o >>= 1) v = fmaxf(v, __shfl_xor(v, o, 32));
  return v;
}
__device__ __forceinline__ float wred_min(float v) {
#pragma unroll
  for (int o = 16; o; o >>= 1) v = fminf(v, __shfl_xor(v, o, 32));
  return v;
}
__device__ __forceinline__ float wred_sum(float v) {
#pragma unroll
  for (int o = 16; o; o >>= 1) v += __shfl_xor(v, o, 32);
  return v;
}
__device__ __forceinline__ int wred_isum(int v) {
#pragma unroll
  for (int o = 16; o; o >>= 1) v += __shfl_xor(v, o, 32);
  return v;
}

// ---------------------------------------------------------------------------
// 1) fp32 -> bf16 convert (flat) and transpose-convert (for WMMA B operands)
// ---------------------------------------------------------------------------
__global__ void k_cvt(bf16* __restrict__ dst, const float* __restrict__ src,
                      size_t n) {
  size_t i = (size_t)blockIdx.x * blockDim.x + threadIdx.x;
  if (i < n) dst[i] = (bf16)src[i];
}

// dst[C x R] = transpose(src[R x C]) in bf16
__global__ void k_cvt_t(bf16* __restrict__ dst, const float* __restrict__ src,
                        int R, int C) {
  size_t i = (size_t)blockIdx.x * blockDim.x + threadIdx.x;
  if (i >= (size_t)R * C) return;
  int n = (int)(i / R);
  int k = (int)(i % R);
  dst[(size_t)n * R + k] = (bf16)src[(size_t)k * C + n];
}

// ---------------------------------------------------------------------------
// 2) QKV projection GEMM: X[4096x1024]bf16 @ {WqT,WkT,WvT} -> QKVf32[4096x2048]
//    block tile 64x128, 8 waves, 32x32 per wave (4 WMMA accumulators)
// ---------------------------------------------------------------------------
__global__ __launch_bounds__(256) void k_gemm_qkv(
    const bf16* __restrict__ Xb, const bf16* __restrict__ WqT,
    const bf16* __restrict__ WkT, const bf16* __restrict__ WvT,
    float* __restrict__ QKV) {
  const int lane = threadIdx.x & 31;
  const int wid  = threadIdx.x >> 5;
  const int m0 = blockIdx.x * 64 + (wid & 1) * 32;
  const int n0 = blockIdx.y * 128 + (wid >> 1) * 32;

  const bf16* bt; int nb;                 // pick weight; 32-col spans never cross
  if (n0 < 1024)      { bt = WqT; nb = n0; }
  else if (n0 < 1536) { bt = WkT; nb = n0 - 1024; }
  else                { bt = WvT; nb = n0 - 1536; }

  v8f acc00 = {}, acc01 = {}, acc10 = {}, acc11 = {};
  for (int k0 = 0; k0 < DD; k0 += 32) {
    v16bf a0 = load_frag_a(Xb, DD, m0,      k0, lane);
    v16bf a1 = load_frag_a(Xb, DD, m0 + 16, k0, lane);
    v16bf b0 = load_frag_b(bt, DD, nb,      k0, lane);
    v16bf b1 = load_frag_b(bt, DD, nb + 16, k0, lane);
    acc00 = WMMA_BF16(a0, b0, acc00);
    acc01 = WMMA_BF16(a0, b1, acc01);
    acc10 = WMMA_BF16(a1, b0, acc10);
    acc11 = WMMA_BF16(a1, b1, acc11);
  }
  store_tile_f32(QKV, NQKV, m0,      n0,      lane, acc00);
  store_tile_f32(QKV, NQKV, m0,      n0 + 16, lane, acc01);
  store_tile_f32(QKV, NQKV, m0 + 16, n0,      lane, acc10);
  store_tile_f32(QKV, NQKV, m0 + 16, n0 + 16, lane, acc11);
}

// ---------------------------------------------------------------------------
// 3) RoPE + bf16 layout: QKVf32 -> Qb[B,NH,L,64], Kb[B,KVH,L,64], VbT[B,KVH,64,L]
// ---------------------------------------------------------------------------
__global__ void k_rope(const float* __restrict__ QKV,
                       const int* __restrict__ pos_ids,
                       bf16* __restrict__ Qb, bf16* __restrict__ Kb,
                       bf16* __restrict__ VbT) {
  size_t i = (size_t)blockIdx.x * blockDim.x + threadIdx.x;
  if (i >= (size_t)MTOT * NQKV) return;
  const int    n  = (int)(i & (NQKV - 1));
  const size_t rm = i >> 11;
  const int b = (int)(rm / LL);
  const int l = (int)(rm % LL);
  const float x = QKV[i];

  if (n < 1536) {                              // Q or K: apply RoPE
    const int d = n & 63;
    const int j = d & 31;
    const float pos  = (float)pos_ids[(size_t)b * LL + l];
    const float freq = pos * __powf(10000.0f, -(float)j * (1.0f / 32.0f));
    const float c = __cosf(freq), s = __sinf(freq);
    const float partner = QKV[rm * NQKV + (size_t)((d < 32) ? n + 32 : n - 32)];
    const float out = (d < 32) ? (x * c - partner * s) : (x * c + partner * s);
    if (n < 1024) {
      const int h = n >> 6;
      Qb[(((size_t)b * NHH + h) * LL + l) * DHH + d] = (bf16)out;
    } else {
      const int kvh = (n - 1024) >> 6;
      Kb[(((size_t)b * KVHH + kvh) * LL + l) * DHH + d] = (bf16)out;
    }
  } else {                                     // V: transpose to [*, 64, L]
    const int n2 = n - 1536;
    const int kvh = n2 >> 6, d = n2 & 63;
    VbT[(((size_t)b * KVHH + kvh) * DHH + d) * LL + l] = (bf16)x;
  }
}

// ---------------------------------------------------------------------------
// 4) Fused attention: per block = (b, h, 16-query tile).
//    Phase 1: scores 16x2048 via WMMA -> LDS (128 KB, fits CDNA5's 320 KB/WGP)
//    Phase 2: exact top-1024 threshold per row (value bisection, wave32)
//    Phase 3: masked softmax -> bf16 probs aliased into each row's LDS slot
//    Phase 4: PV 16x64 via WMMA (A from LDS via ds_load, B from L2-resident V^T)
// ---------------------------------------------------------------------------
__global__ __launch_bounds__(256) void k_attn(const bf16* __restrict__ Qb,
                                              const bf16* __restrict__ Kb,
                                              const bf16* __restrict__ VbT,
                                              bf16* __restrict__ Attnb) {
  extern __shared__ __align__(16) char smem_raw[];
  float* sc = (float*)smem_raw;                 // [16][2048] fp32 scores

  const int lane = threadIdx.x & 31;
  const int wid  = threadIdx.x >> 5;
  const int qt = blockIdx.x & 127;              // query tile (L/16 = 128)
  const int h  = (blockIdx.x >> 7) & 15;
  const int b  = blockIdx.x >> 11;
  const int kvh = h / GRP;                      // torch-style interleaved repeat_kv

  const bf16* Qh  = Qb  + (((size_t)b * NHH  + h)   * LL + (size_t)qt * 16) * DHH;
  const bf16* Kh  = Kb  + (((size_t)b * KVHH + kvh) * LL) * DHH;
  const bf16* VhT = VbT + (((size_t)b * KVHH + kvh) * DHH) * (size_t)LL;

  // ---- Phase 1: scores = (Q K^T) * DH^-0.5 --------------------------------
  const v16bf aq0 = load_frag_a(Qh, DHH, 0, 0,  lane);   // reused across tiles
  const v16bf aq1 = load_frag_a(Qh, DHH, 0, 32, lane);
  for (int t = wid; t < LL / 16; t += 8) {               // 16 key tiles / wave
    const int kn0 = t * 16;
    v8f acc = {};
    acc = WMMA_BF16(aq0, load_frag_b(Kh, DHH, kn0, 0,  lane), acc);
    acc = WMMA_BF16(aq1, load_frag_b(Kh, DHH, kn0, 32, lane), acc);
#pragma unroll
    for (int r = 0; r < 8; ++r) acc[r] *= 0.125f;        // 64^-0.5
    store_tile_f32(sc, LL, 0, kn0, lane, acc);
  }
  __syncthreads();

  // ---- Phase 2+3: per-row exact-k threshold + softmax (2 rows per wave) ---
  for (int r = wid; r < 16; r += 8) {
    float* row = sc + (size_t)r * LL;
    float mx = -3.4e38f, mn = 3.4e38f;
    for (int i = lane; i < LL; i += 32) {
      const float v = row[i];
      mx = fmaxf(mx, v);
      mn = fminf(mn, v);
    }
    mx = wred_max(mx);
    mn = wred_min(mn);

    float lo = mn, hi = mx;                    // largest t with count(>=t) >= k
    for (int it = 0; it < 24; ++it) {
      const float mid = 0.5f * (lo + hi);
      int c = 0;
      for (int i = lane; i < LL; i += 32) c += (row[i] >= mid);
      c = wred_isum(c);
      if (c >= TOPK) lo = mid; else hi = mid;
    }
    const float thr = lo;

    float s = 0.0f;
    for (int i = lane; i < LL; i += 32) {
      const float v = row[i];
      if (v >= thr) s += __expf(v - mx);
    }
    s = wred_sum(s);
    const float inv = 1.0f / s;

    // probs (bf16) aliased over the first half of this row's own fp32 slot;
    // all reads of `row` by this wave are complete before the writes below.
    bf16* prow = (bf16*)row;
    for (int i = lane; i < LL; i += 32) {
      const float v = row[i];
      const bf16 p = (bf16)((v >= thr) ? __expf(v - mx) * inv : 0.0f);
      __syncwarp();                            // lockstep wave32: reads precede writes
      prow[i] = p;
    }
  }
  __syncthreads();

  // ---- Phase 4: out = probs @ V  (waves 0-3, one 16x16 d-tile each) -------
  if (wid < 4) {
    const int n0 = wid * 16;                   // d-range of this tile
    const bf16* pr = (const bf16*)sc;          // lda = 4096 bf16 (8 KB row pitch)
    v8f acc = {};
    for (int k0 = 0; k0 < LL; k0 += 32) {
      v16bf ap = load_frag_a(pr, 2 * LL, 0, k0, lane);     // ds_load from LDS
      v16bf bv = load_frag_b(VhT, LL, n0, k0, lane);       // V^T rows = d
      acc = WMMA_BF16(ap, bv, acc);
    }
    const int m = ((lane >> 4) << 3);
    const int n = n0 + (lane & 15);
#pragma unroll
    for (int r = 0; r < 8; ++r) {
      Attnb[((size_t)b * LL + (size_t)qt * 16 + m + r) * (NHH * DHH) +
            (size_t)h * DHH + n] = (bf16)acc[r];
    }
  }
}

// ---------------------------------------------------------------------------
// 5) Output projection: Attnb[4096x1024]bf16 @ WoT -> d_out fp32
// ---------------------------------------------------------------------------
__global__ __launch_bounds__(256) void k_gemm_out(const bf16* __restrict__ Ab,
                                                  const bf16* __restrict__ WoT,
                                                  float* __restrict__ Out) {
  const int lane = threadIdx.x & 31;
  const int wid  = threadIdx.x >> 5;
  const int m0 = blockIdx.x * 64 + (wid & 1) * 32;
  const int n0 = blockIdx.y * 128 + (wid >> 1) * 32;

  v8f acc00 = {}, acc01 = {}, acc10 = {}, acc11 = {};
  for (int k0 = 0; k0 < NHH * DHH; k0 += 32) {
    v16bf a0 = load_frag_a(Ab, NHH * DHH, m0,      k0, lane);
    v16bf a1 = load_frag_a(Ab, NHH * DHH, m0 + 16, k0, lane);
    v16bf b0 = load_frag_b(WoT, NHH * DHH, n0,      k0, lane);
    v16bf b1 = load_frag_b(WoT, NHH * DHH, n0 + 16, k0, lane);
    acc00 = WMMA_BF16(a0, b0, acc00);
    acc01 = WMMA_BF16(a0, b1, acc01);
    acc10 = WMMA_BF16(a1, b0, acc10);
    acc11 = WMMA_BF16(a1, b1, acc11);
  }
  store_tile_f32(Out, DD, m0,      n0,      lane, acc00);
  store_tile_f32(Out, DD, m0,      n0 + 16, lane, acc01);
  store_tile_f32(Out, DD, m0 + 16, n0,      lane, acc10);
  store_tile_f32(Out, DD, m0 + 16, n0 + 16, lane, acc11);
}

// ---------------------------------------------------------------------------
// Host launcher. Inputs: hidden, wq, wk, wv, wo, position_ids.
// ---------------------------------------------------------------------------
extern "C" void kernel_launch(void* const* d_in, const int* in_sizes, int n_in,
                              void* d_out, int out_size, void* d_ws,
                              size_t ws_size, hipStream_t stream) {
  const float* hid = (const float*)d_in[0];
  const float* wq  = (const float*)d_in[1];
  const float* wk  = (const float*)d_in[2];
  const float* wv  = (const float*)d_in[3];
  const float* wo  = (const float*)d_in[4];
  const int*   pid = (const int*)d_in[5];
  float* out = (float*)d_out;

  // workspace carve-up (bytes, 256-aligned) — ~73 MB total
  char* ws = (char*)d_ws;
  size_t o = 0;
  auto take = [&](size_t bytes) { char* p = ws + o; o = (o + bytes + 255) & ~(size_t)255; return p; };
  bf16*  Xb    = (bf16*)take((size_t)MTOT * DD * 2);           // 8 MB
  bf16*  WqT   = (bf16*)take((size_t)1024 * DD * 2);           // 2 MB
  bf16*  WkT   = (bf16*)take((size_t)512  * DD * 2);           // 1 MB
  bf16*  WvT   = (bf16*)take((size_t)512  * DD * 2);           // 1 MB
  bf16*  WoT   = (bf16*)take((size_t)DD * 1024 * 2);           // 2 MB
  float* QKVf  = (float*)take((size_t)MTOT * NQKV * 4);        // 32 MB
  bf16*  Qb    = (bf16*)take((size_t)BB * NHH  * LL * DHH * 2);// 8 MB
  bf16*  Kb    = (bf16*)take((size_t)BB * KVHH * LL * DHH * 2);// 4 MB
  bf16*  VbT   = (bf16*)take((size_t)BB * KVHH * DHH * LL * 2);// 4 MB
  bf16*  Attnb = (bf16*)take((size_t)MTOT * NHH * DHH * 2);    // 8 MB
  (void)ws_size; (void)n_in; (void)in_sizes; (void)out_size;

  // 1) convert / transpose
  {
    size_t n = (size_t)MTOT * DD;
    k_cvt<<<dim3((unsigned)((n + 255) / 256)), dim3(256), 0, stream>>>(Xb, hid, n);
  }
  k_cvt_t<<<dim3((1024u * 1024u + 255) / 256), dim3(256), 0, stream>>>(WqT, wq, DD, 1024);
  k_cvt_t<<<dim3((1024u * 512u  + 255) / 256), dim3(256), 0, stream>>>(WkT, wk, DD, 512);
  k_cvt_t<<<dim3((1024u * 512u  + 255) / 256), dim3(256), 0, stream>>>(WvT, wv, DD, 512);
  k_cvt_t<<<dim3((1024u * 1024u + 255) / 256), dim3(256), 0, stream>>>(WoT, wo, 1024, DD);

  // 2) QKV projection (M=4096, N=2048, K=1024)
  k_gemm_qkv<<<dim3(MTOT / 64, NQKV / 128), dim3(256), 0, stream>>>(Xb, WqT, WkT, WvT, QKVf);

  // 3) RoPE + bf16 layouts
  {
    size_t n = (size_t)MTOT * NQKV;
    k_rope<<<dim3((unsigned)((n + 255) / 256)), dim3(256), 0, stream>>>(QKVf, pid, Qb, Kb, VbT);
  }

  // 4) fused scores -> top-k -> softmax -> PV  (4096 blocks, 128 KB dynamic LDS)
  k_attn<<<dim3(BB * NHH * (LL / 16)), dim3(256), 16 * LL * 4, stream>>>(Qb, Kb, VbT, Attnb);

  // 5) output projection (M=4096, N=1024, K=1024)
  k_gemm_out<<<dim3(MTOT / 64, DD / 128), dim3(256), 0, stream>>>(Attnb, WoT, out);
}
